// MultiScreen_34497177321841
// MI455X (gfx1250) — matrix operand order repaired and checked by reading, not compiled, not run on hardware
//
#include <hip/hip_runtime.h>
#include <hip/hip_bf16.h>
#include <math.h>

typedef __bf16 bf16_t;
typedef __attribute__((ext_vector_type(16))) __bf16 v16bf;
typedef __attribute__((ext_vector_type(8)))  float   v8f;

#define N_HEADS   16
#define D_KEY     16
#define D_VAL     64
#define N_TOK     2048
#define N_BATCH   4
#define DIM       1024
#define DIM_INNER 1280
#define N_ROWS    8192
#define EPSF      1e-12f

union FragBF { v16bf v; uint4 u[2]; };

__device__ __forceinline__ v8f wmma_bf16(v16bf a, v16bf b, v8f c) {
    // D = A(16x32 bf16) x B(32x16 bf16) + C(16x16 f32)
    return __builtin_amdgcn_wmma_f32_16x16x32_bf16(false, a, false, b,
                                                   (short)0, c, false, false);
}

// A fragment (16x32 bf16) from LDS strip laid row-major [rows][32 bf16].
// Per ISA: regs0-3 hold k = hi*8 + {0..7}, regs4-7 hold k = 16 + hi*8 + {0..7}.
__device__ __forceinline__ v16bf load_fragA(const uint* lds, int baseDw, int row0) {
    const int lane = threadIdx.x & 31;
    const int hi   = lane >> 4;
    FragBF f;
    const uint4* p = (const uint4*)(lds + baseDw + (row0 + (lane & 15)) * 16);
    f.u[0] = p[hi];
    f.u[1] = p[2 + hi];
    return f.v;
}

// B fragment (32x16 bf16) from LDS tile stored TRANSPOSED: [n][32 k] row-major.
// Per ISA: lane n = lane&15, regs0-3 hold k = hi*8 + {0..7}, regs4-7 hold k = 16 + hi*8 + {0..7}.
__device__ __forceinline__ v16bf load_fragB_T(const uint* lds, int baseDw, int n0) {
    const int lane = threadIdx.x & 31;
    const int hi   = lane >> 4;
    FragBF f;
    const uint4* p = (const uint4*)(lds + baseDw + (n0 + (lane & 15)) * 16);
    f.u[0] = p[hi];
    f.u[1] = p[2 + hi];
    return f.v;
}

// Cheap tanh built on native v_exp_f32; saturates correctly at +/-inf.
__device__ __forceinline__ float fast_tanh(float x) {
    float e = __expf(2.0f * x);
    return 1.0f - 2.0f / (e + 1.0f);
}

// ---------------------------------------------------------------- stage 0
__global__ __launch_bounds__(256) void cvt_f32_bf16(const float* __restrict__ src,
                                                    bf16_t* __restrict__ dst, int n) {
    int stride = gridDim.x * blockDim.x;
    for (int i = blockIdx.x * blockDim.x + threadIdx.x; i < n; i += stride)
        dst[i] = (bf16_t)src[i];
}

// ---------------------------------------------------------------- stage 1
// block = 256 token-rows x one head x (qg|kv). Wave = 32 rows x 80 cols
// (2 A-frags x 5 B-frags -> 10 WMMA/iter). Epilogue: C tiles -> LDS, then
// thread-per-row (contiguous) l2-norm / gate activation, coalesced stores.
__global__ __launch_bounds__(256, 1) void qkv_gemm_kernel(
    const bf16_t* __restrict__ tokensB,
    const bf16_t* __restrict__ WqgB, const bf16_t* __restrict__ WkvB,
    bf16_t* __restrict__ qn, bf16_t* __restrict__ kn,
    bf16_t* __restrict__ vn, float* __restrict__ gate)
{
    // Pool: loop phase uses [0,8192) A-tiles + [8192,9472) B-tiles.
    // Epilogue phase reuses [0,10368) as float[128][81] (padded, bank-clean).
    __shared__ __align__(16) uint pool[10368];   // 41.5 KB
    uint*  ldsA = pool;
    uint*  ldsB = pool + 8192;
    float* ldsE = (float*)pool;

    const int tid = threadIdx.x, waveId = tid >> 5, lane = tid & 31;
    const int ln = lane & 15, hi = lane >> 4;
    const int rowBase = blockIdx.x * 256;
    const int head    = blockIdx.y;
    const int which   = blockIdx.z;            // 0 = qg, 1 = kv
    const int colBase = head * 80;
    const bf16_t* W = which ? WkvB : WqgB;

    v8f acc[2][5];
    for (int g = 0; g < 2; ++g)
        for (int t = 0; t < 5; ++t)
            for (int e = 0; e < 8; ++e) acc[g][t][e] = 0.0f;

    const uint4* gA4 = (const uint4*)tokensB;
    uint4* ldsA4 = (uint4*)ldsA;
    bf16_t* ldsBh = (bf16_t*)ldsB;

    for (int kb = 0; kb < DIM; kb += 32) {
        #pragma unroll
        for (int i = 0; i < 8; ++i) {          // A: 2048 uint4 (b128 loads)
            int slot = tid + i * 256;
            int row = slot >> 2, q = slot & 3;
            ldsA4[slot] = gA4[(size_t)(rowBase + row) * (DIM / 8) + (kb >> 3) + q];
        }
        #pragma unroll
        for (int i = 0; i < 10; ++i) {         // B: 32x80 bf16 -> [n][k]
            int e = tid + i * 256;
            int k = e / 80, n = e - k * 80;
            ldsBh[n * 32 + k] = W[(size_t)(kb + k) * DIM_INNER + colBase + n];
        }
        __syncthreads();
        v16bf a0 = load_fragA(ldsA, 0, waveId * 32);
        v16bf a1 = load_fragA(ldsA, 0, waveId * 32 + 16);
        #pragma unroll
        for (int t = 0; t < 5; ++t) {
            v16bf b = load_fragB_T(ldsB, 0, t * 16);
            acc[0][t] = wmma_bf16(a0, b, acc[0][t]);
            acc[1][t] = wmma_bf16(a1, b, acc[1][t]);
        }
        __syncthreads();
    }

    // Epilogue: one 128-row pass per row-group.
    for (int g = 0; g < 2; ++g) {
        __syncthreads();
        #pragma unroll
        for (int t = 0; t < 5; ++t)
            #pragma unroll
            for (int r = 0; r < 8; ++r)
                ldsE[(waveId * 16 + r + 8 * hi) * 81 + t * 16 + ln] = acc[g][t][r];
        __syncthreads();
        if (tid < 128) {
            const int wv = tid >> 4, rl = tid & 15;
            const int grow = rowBase + wv * 32 + g * 16 + rl;
            const int batch = grow >> 11, tok = grow & (N_TOK - 1);
            const float* rowp = ldsE + tid * 81;
            const size_t hb = (size_t)(batch * N_HEADS + head) * N_TOK + tok;
            // cols 0..15: q or k, l2-norm over 16 dims
            float s = 0.f;
            #pragma unroll
            for (int c = 0; c < 16; ++c) { float x = rowp[c]; s += x * x; }
            float inv = 1.0f / fmaxf(sqrtf(s), EPSF);
            bf16_t* dqk = (which ? kn : qn) + hb * D_KEY;
            #pragma unroll
            for (int c = 0; c < 16; ++c) dqk[c] = (bf16_t)(rowp[c] * inv);
            if (which == 0) {      // cols 16..79: gates tanh(silu(x))
                float* gd = gate + hb * D_VAL;
                for (int c = 0; c < 64; ++c) {
                    float x = rowp[16 + c];
                    gd[c] = fast_tanh(x / (1.0f + __expf(-x)));
                }
            } else {               // cols 16..79: v, l2-norm over 64 dims
                float sv = 0.f;
                for (int c = 0; c < 64; ++c) { float x = rowp[16 + c]; sv += x * x; }
                float invv = 1.0f / fmaxf(sqrtf(sv), EPSF);
                bf16_t* vd = vn + hb * D_VAL;
                for (int c = 0; c < 64; ++c) vd[c] = (bf16_t)(rowp[16 + c] * invv);
            }
        }
    }
}

// ---------------------------------------------------------------- stage 2
// block = one (batch,head) x 256 query rows; wave = 32 q rows; streams K/V in
// 32-key blocks. relu^2 tiles bounce through per-wave LDS to become A-frags.
__global__ __launch_bounds__(256, 1) void attn_kernel(
    const bf16_t* __restrict__ qn, const bf16_t* __restrict__ kn,
    const bf16_t* __restrict__ vn, const float* __restrict__ gate,
    bf16_t* __restrict__ attnOut)
{
    // Pool: loop uses K[0,256) + V[256,1280) + attn scratch [1280,5376).
    // Epilogue reuses [0,8320) as float[128][65].
    __shared__ __align__(16) uint pool[8320];    // 33.3 KB
    uint*  ldsK    = pool;
    uint*  ldsV    = pool + 256;
    uint*  ldsAttn = pool + 1280;
    float* ldsE    = (float*)pool;

    const int tid = threadIdx.x, waveId = tid >> 5, lane = tid & 31;
    const int ln = lane & 15, hi = lane >> 4;
    const int bh = blockIdx.y, batch = bh >> 4, head = bh & 15;
    const int qBase = blockIdx.x * 256 + waveId * 32;
    const size_t seqBase = (size_t)bh * N_TOK;

    // Q fragments, constant over the key loop; K-dim padded 16 -> 32 with zeros.
    FragBF aq[2];
    #pragma unroll
    for (int g = 0; g < 2; ++g) {
        const uint4* qp = (const uint4*)(qn + (seqBase + qBase + g * 16 + ln) * D_KEY);
        aq[g].u[0] = qp[hi];
        aq[g].u[1].x = aq[g].u[1].y = aq[g].u[1].z = aq[g].u[1].w = 0u;
    }

    v8f acc[2][4];
    for (int g = 0; g < 2; ++g)
        for (int t = 0; t < 4; ++t)
            for (int e = 0; e < 8; ++e) acc[g][t][e] = 0.f;

    bf16_t* ldsVh = (bf16_t*)ldsV;
    bf16_t* attnScratch = (bf16_t*)(ldsAttn + waveId * 512);

    for (int j = 0; j < N_TOK; j += 32) {
        {   // stage K block (256 dwords, 1/thread)
            int row = tid >> 3, cdw = tid & 7;
            ldsK[tid] = ((const uint*)kn)[(seqBase + j + row) * 8 + cdw];
        }
        #pragma unroll
        for (int i = 0; i < 8; ++i) {  // stage V block 32x64 -> [n][k]
            int e = tid + i * 256;
            int k = e >> 6, n = e & 63;
            ldsVh[n * 32 + k] = vn[(seqBase + j + k) * D_VAL + n];
        }
        __syncthreads();

        // sim tiles, relu^2, repack C-layout -> A-layout via per-wave LDS.
        #pragma unroll
        for (int t = 0; t < 2; ++t) {
            FragBF bk;
            {
                const uint4* p = (const uint4*)ldsK + (t * 16 + ln) * 2;
                bk.u[0] = p[hi];
                bk.u[1].x = bk.u[1].y = bk.u[1].z = bk.u[1].w = 0u;
            }
            v8f z; for (int e = 0; e < 8; ++e) z[e] = 0.f;
            #pragma unroll
            for (int g = 0; g < 2; ++g) {
                v8f sim = wmma_bf16(aq[g].v, bk.v, z);
                #pragma unroll
                for (int r = 0; r < 8; ++r) {
                    float x = fmaxf(sim[r], 0.f);
                    attnScratch[g * 512 + (r + 8 * hi) * 32 + t * 16 + ln] =
                        (bf16_t)(x * x);
                }
            }
        }
        // aggr += attn(16x32) @ V(32x64). Same-wave LDS ops are in-order.
        v16bf aA0 = load_fragA(ldsAttn, waveId * 512, 0);
        v16bf aA1 = load_fragA(ldsAttn, waveId * 512 + 256, 0);
        #pragma unroll
        for (int t = 0; t < 4; ++t) {
            v16bf bv = load_fragB_T(ldsV, 0, t * 16);
            acc[0][t] = wmma_bf16(aA0, bv, acc[0][t]);
            acc[1][t] = wmma_bf16(aA1, bv, acc[1][t]);
        }
        __syncthreads();
    }

    // Epilogue: tanh-norm over 64 dims * gate; thread-per-row, coalesced.
    for (int g = 0; g < 2; ++g) {
        __syncthreads();
        #pragma unroll
        for (int t = 0; t < 4; ++t)
            #pragma unroll
            for (int r = 0; r < 8; ++r)
                ldsE[(waveId * 16 + r + 8 * hi) * 65 + t * 16 + ln] = acc[g][t][r];
        __syncthreads();
        if (tid < 128) {
            const int wv = tid >> 4, rl = tid & 15;
            const int tok = blockIdx.x * 256 + wv * 32 + g * 16 + rl;
            const float* rowp = ldsE + tid * 65;
            float s = 0.f;
            for (int c = 0; c < 64; ++c) { float x = rowp[c]; s += x * x; }
            float nrm = sqrtf(s);
            float factor = fast_tanh(nrm) / fmaxf(nrm, EPSF);
            const float* gp = gate + (seqBase + tok) * D_VAL;
            bf16_t* op = attnOut + (size_t)(batch * N_TOK + tok) * DIM + head * D_VAL;
            for (int c = 0; c < 64; ++c)
                op[c] = (bf16_t)(rowp[c] * factor * gp[c]);
        }
    }
}

// ---------------------------------------------------------------- stage 3
// out[8192x1024] = attnOut(bf16) @ Wout(bf16), f32 stores. Wave = 32x64.
__global__ __launch_bounds__(256, 1) void out_gemm_kernel(
    const bf16_t* __restrict__ attnOut, const bf16_t* __restrict__ WoutB,
    float* __restrict__ out)
{
    // Pool: loop uses A[0,8192) + B[8192,9216). Epilogue reuses float[128][65].
    __shared__ __align__(16) uint pool[9216];    // 36.9 KB
    uint*  ldsA = pool;
    uint*  ldsB = pool + 8192;
    float* ldsE = (float*)pool;

    const int tid = threadIdx.x, waveId = tid >> 5, lane = tid & 31;
    const int ln = lane & 15, hi = lane >> 4;
    const int rowBase = blockIdx.x * 256;
    const int colBase = blockIdx.y * 64;

    v8f acc[2][4];
    for (int g = 0; g < 2; ++g)
        for (int t = 0; t < 4; ++t)
            for (int e = 0; e < 8; ++e) acc[g][t][e] = 0.f;

    const uint4* gA4 = (const uint4*)attnOut;
    uint4* ldsA4 = (uint4*)ldsA;
    bf16_t* ldsBh = (bf16_t*)ldsB;

    for (int kb = 0; kb < DIM; kb += 32) {
        #pragma unroll
        for (int i = 0; i < 8; ++i) {
            int slot = tid + i * 256;
            int row = slot >> 2, q = slot & 3;
            ldsA4[slot] = gA4[(size_t)(rowBase + row) * (DIM / 8) + (kb >> 3) + q];
        }
        #pragma unroll
        for (int i = 0; i < 8; ++i) {
            int e = tid + i * 256;
            int k = e >> 6, n = e & 63;
            ldsBh[n * 32 + k] = WoutB[(size_t)(kb + k) * DIM + colBase + n];
        }
        __syncthreads();
        v16bf a0 = load_fragA(ldsA, 0, waveId * 32);
        v16bf a1 = load_fragA(ldsA, 0, waveId * 32 + 16);
        #pragma unroll
        for (int t = 0; t < 4; ++t) {
            v16bf b = load_fragB_T(ldsB, 0, t * 16);
            acc[0][t] = wmma_bf16(a0, b, acc[0][t]);
            acc[1][t] = wmma_bf16(a1, b, acc[1][t]);
        }
        __syncthreads();
    }

    for (int g = 0; g < 2; ++g) {
        __syncthreads();
        #pragma unroll
        for (int t = 0; t < 4; ++t)
            #pragma unroll
            for (int r = 0; r < 8; ++r)
                ldsE[(waveId * 16 + r + 8 * hi) * 65 + t * 16 + ln] = acc[g][t][r];
        __syncthreads();
        if (tid < 128) {
            const int row = rowBase + (tid >> 4) * 32 + g * 16 + (tid & 15);
            float* op = out + (size_t)row * DIM + colBase;
            const float* rowp = ldsE + tid * 65;
            for (int c = 0; c < 64; ++c) op[c] = rowp[c];
        }
    }
}

// ---------------------------------------------------------------- launcher
extern "C" void kernel_launch(void* const* d_in, const int* in_sizes, int n_in,
                              void* d_out, int out_size, void* d_ws, size_t ws_size,
                              hipStream_t stream) {
    (void)in_sizes; (void)n_in; (void)out_size; (void)ws_size;
    const float* tokens = (const float*)d_in[0];
    const float* Wqg    = (const float*)d_in[1];
    const float* Wkv    = (const float*)d_in[2];
    const float* Wout   = (const float*)d_in[3];
    float* out = (float*)d_out;

    char* ws = (char*)d_ws;
    size_t off = 0;
    auto alloc = [&](size_t bytes) {
        void* p = ws + off;
        off = (off + bytes + 255) & ~(size_t)255;
        return p;
    };
    bf16_t* tokensB = (bf16_t*)alloc((size_t)N_ROWS * DIM * 2);
    bf16_t* WqgB    = (bf16_t*)alloc((size_t)DIM * DIM_INNER * 2);
    bf16_t* WkvB    = (bf16_t*)alloc((size_t)DIM * DIM_INNER * 2);
    bf16_t* WoutB   = (bf16_t*)alloc((size_t)DIM * DIM * 2);
    bf16_t* qn      = (bf16_t*)alloc((size_t)N_BATCH * N_HEADS * N_TOK * D_KEY * 2);
    bf16_t* kn      = (bf16_t*)alloc((size_t)N_BATCH * N_HEADS * N_TOK * D_KEY * 2);
    bf16_t* vn      = (bf16_t*)alloc((size_t)N_BATCH * N_HEADS * N_TOK * D_VAL * 2);
    float*  gate    = (float*)alloc((size_t)N_BATCH * N_HEADS * N_TOK * D_VAL * 4);
    bf16_t* attnOut = (bf16_t*)alloc((size_t)N_ROWS * DIM * 2);

    cvt_f32_bf16<<<1024, 256, 0, stream>>>(tokens, tokensB, N_ROWS * DIM);
    cvt_f32_bf16<<<1024, 256, 0, stream>>>(Wqg, WqgB, DIM * DIM_INNER);
    cvt_f32_bf16<<<1024, 256, 0, stream>>>(Wkv, WkvB, DIM * DIM_INNER);
    cvt_f32_bf16<<<1024, 256, 0, stream>>>(Wout, WoutB, DIM * DIM);

    qkv_gemm_kernel<<<dim3(N_ROWS / 256, N_HEADS, 2), 256, 0, stream>>>(
        tokensB, WqgB, WkvB, qn, kn, vn, gate);

    attn_kernel<<<dim3(N_TOK / 256, N_BATCH * N_HEADS), 256, 0, stream>>>(
        qn, kn, vn, gate, attnOut);

    out_gemm_kernel<<<dim3(N_ROWS / 256, DIM / 64), 256, 0, stream>>>(
        attnOut, WoutB, out);
}
